// GCNModel_40888088658484
// MI455X (gfx1250) — matrix operand order repaired and checked by reading, not compiled, not run on hardware
//
#include <hip/hip_runtime.h>
#include <hip/hip_bf16.h>

#define NROWS  8192
#define NFEATC 512
#define NHIDC  128
#define NCLASSC 40
#define LAM0C  1.0f
#define LAM1C  1.0f
#define ALPHAC (1.0f / 3.0f)
#define STEPSC 8
#define KCHUNKS 4
#define KLEN   (NROWS / KCHUNKS)   // 2048
#define KITERS (KLEN / 32)         // 64

// LDS B tile: 128 rows x 32 bf16, padded row stride 40 bf16 (80B) for banks
#define BK     32
#define BPAD   40
#define BUFELT (NHIDC * BPAD)      // 5120 bf16 = 10240 B per buffer

typedef __bf16 bf16;
typedef __attribute__((ext_vector_type(4)))  __bf16 v4bf;
typedef __attribute__((ext_vector_type(8)))  __bf16 v8bf;
typedef __attribute__((ext_vector_type(16))) __bf16 v16bf;
typedef __attribute__((ext_vector_type(2)))  float  v2f;
typedef __attribute__((ext_vector_type(4)))  float  v4f;
typedef __attribute__((ext_vector_type(8)))  float  v8f;
typedef __attribute__((ext_vector_type(4)))  unsigned int u32x4;
typedef __attribute__((ext_vector_type(8)))  int i32x8;
typedef __attribute__((ext_vector_type(4)))  int i32x4;

#if defined(__has_builtin)
# if __has_builtin(__builtin_amdgcn_wmma_f32_16x16x4_f32)
#  define HAVE_WMMA_F32X4 1
# endif
# if __has_builtin(__builtin_amdgcn_tensor_load_to_lds)
#  define HAVE_TDM 1
# endif
#endif
#ifndef HAVE_WMMA_F32X4
# define HAVE_WMMA_F32X4 0
#endif
#ifndef HAVE_TDM
# define HAVE_TDM 0
#endif

union ABFrag { v16bf v; v8bf h[2]; };

// ---------------------------------------------------------------------------
// One-time: fp32 A matrices -> bf16 (halves HBM traffic for the 16 big GEMMs)
// ---------------------------------------------------------------------------
__global__ __launch_bounds__(256) void k_convertA(const float* __restrict__ Ab,
                                                  const float* __restrict__ Ag,
                                                  bf16* __restrict__ AbB,
                                                  bf16* __restrict__ AgB) {
    size_t t    = (size_t)blockIdx.x * blockDim.x + threadIdx.x;
    size_t base = t * 8;
    const size_t half = (size_t)NROWS * NROWS;
    const float* src;
    bf16*        dst;
    if (base < half) { src = Ab + base;          dst = AbB + base; }
    else             { src = Ag + (base - half); dst = AgB + (base - half); }
    v4f f0 = *(const v4f*)(src);
    v4f f1 = *(const v4f*)(src + 4);
    v8bf o;
    o[0] = (__bf16)f0[0]; o[1] = (__bf16)f0[1]; o[2] = (__bf16)f0[2]; o[3] = (__bf16)f0[3];
    o[4] = (__bf16)f1[0]; o[5] = (__bf16)f1[1]; o[6] = (__bf16)f1[2]; o[7] = (__bf16)f1[3];
    *(v8bf*)dst = o;
}

// ---------------------------------------------------------------------------
// One-time: S1=H1+H1^T, P1=H1@H1^T, S2, P2 (all symmetric) and q_inv vector
// ---------------------------------------------------------------------------
__global__ __launch_bounds__(256) void k_prep(const float* __restrict__ H1,
                                              const float* __restrict__ H2,
                                              const float* __restrict__ db,
                                              const float* __restrict__ dg,
                                              float* __restrict__ S1, float* __restrict__ P1,
                                              float* __restrict__ S2, float* __restrict__ P2,
                                              float* __restrict__ qinv) {
    int gid = blockIdx.x * 256 + threadIdx.x;
    if (gid < NHIDC * NHIDC) {
        int i = gid >> 7, j = gid & 127;
        const float* h1i = H1 + (size_t)i * NHIDC;
        const float* h1j = H1 + (size_t)j * NHIDC;
        const float* h2i = H2 + (size_t)i * NHIDC;
        const float* h2j = H2 + (size_t)j * NHIDC;
        float p1 = 0.f, p2 = 0.f;
        for (int k = 0; k < NHIDC; ++k) {
            p1 = fmaf(h1i[k], h1j[k], p1);
            p2 = fmaf(h2i[k], h2j[k], p2);
        }
        S1[gid] = H1[(size_t)i * NHIDC + j] + H1[(size_t)j * NHIDC + i];
        S2[gid] = H2[(size_t)i * NHIDC + j] + H2[(size_t)j * NHIDC + i];
        P1[gid] = p1;
        P2[gid] = p2;
    } else {
        int q = gid - NHIDC * NHIDC;
        if (q < NROWS) qinv[q] = 1.0f / (LAM0C * db[q] + LAM1C * dg[q] + 1.0f);
    }
}

// ---------------------------------------------------------------------------
// One-time: Y0 = relu(x @ W1^T + b1).  x row staged in LDS; W1 is L2-resident.
// ---------------------------------------------------------------------------
__global__ __launch_bounds__(256) void k_y0(const float* __restrict__ x,
                                            const float* __restrict__ W1,
                                            const float* __restrict__ b1,
                                            float* __restrict__ Y0) {
    __shared__ float xs[2][NFEATC];
    int tid = threadIdx.x;
    for (int i = tid; i < 2 * NFEATC; i += 256) {
        int r = i >> 9, c = i & 511;
        xs[r][c] = x[(size_t)(blockIdx.x * 2 + r) * NFEATC + c];
    }
    __syncthreads();
    int mrow = blockIdx.x * 2 + (tid >> 7);
    int h    = tid & 127;
    const float* xr = xs[tid >> 7];
    const float* w  = W1 + (size_t)h * NFEATC;
    float acc = b1[h];
    for (int k = 0; k < NFEATC; ++k) acc = fmaf(xr[k], w[k], acc);
    Y0[(size_t)mrow * NHIDC + h] = fmaxf(acc, 0.0f);
}

// ---------------------------------------------------------------------------
// Per step: Y fp32 [8192][128] -> Yt bf16 [128][8192] (transposed, LDS-tiled)
// so big-GEMM B fragments become contiguous loads.
// ---------------------------------------------------------------------------
__global__ __launch_bounds__(256) void k_transpose(const float* __restrict__ Y,
                                                   bf16* __restrict__ Yt) {
    __shared__ float tile[32][33];
    int tid = threadIdx.x;
    int r  = tid >> 3;        // 0..31
    int c4 = (tid & 7) * 4;   // 0,4,...,28
    int bm = blockIdx.x;      // 0..255
    int bn = blockIdx.y;      // 0..3
    v4f v = *(const v4f*)(Y + (size_t)(bm * 32 + r) * NHIDC + bn * 32 + c4);
    tile[r][c4 + 0] = v[0]; tile[r][c4 + 1] = v[1];
    tile[r][c4 + 2] = v[2]; tile[r][c4 + 3] = v[3];
    __syncthreads();
    v4bf o;
    o[0] = (__bf16)tile[c4 + 0][r];
    o[1] = (__bf16)tile[c4 + 1][r];
    o[2] = (__bf16)tile[c4 + 2][r];
    o[3] = (__bf16)tile[c4 + 3][r];
    *(v4bf*)(Yt + (size_t)(bn * 32 + r) * NROWS + bm * 32 + c4) = o;
}

// ---------------------------------------------------------------------------
// Per step big GEMM: partial[kc] = A_bf16[:, kc] @ Y[kc, :] for both A
// matrices via v_wmma_f32_16x16x32_bf16.
//   - Block = 8 waves, one (mat,kchunk,128-row group); waves share a
//     double-buffered 128x32 bf16 B tile in LDS (TDM-filled, padded stride).
//   - k-loop manually unrolled x2 with explicit a0/a1 A-fragment ping-pong
//     (no register copies), and the 8 N-tiles consumed in two groups of 4 to
//     keep VGPR pressure low enough that accumulators stay pinned (avoids
//     acc shuffling + WMMA->VALU hazard NOPs seen in earlier codegen).
// ---------------------------------------------------------------------------
__global__ __launch_bounds__(256) void k_biggemm(const bf16* __restrict__ Ab,
                                                 const bf16* __restrict__ Ag,
                                                 const bf16* __restrict__ Yt,
                                                 float* __restrict__ Pab,
                                                 float* __restrict__ Pag) {
    __shared__ bf16 Bt[2][BUFELT];
    int tid  = threadIdx.x;
    int widx = tid >> 5;            // wave in block: 0..7
    int lane = tid & 31;
    int blk  = blockIdx.x;          // 0..511
    int mat    = blk >> 8;          // 0: beta, 1: gamma
    int kchunk = (blk >> 6) & 3;    // 0..3
    int mgroup = blk & 63;          // 0..63 -> 128-row group
    int mstrip = (mgroup << 3) + widx;
    const bf16* A = mat ? Ag : Ab;
    float*      C = (mat ? Pag : Pab) + (size_t)kchunk * NROWS * NHIDC;
    int m0  = mstrip << 4;
    int l15 = lane & 15;
    int hi  = lane >> 4;
    int k0  = kchunk * KLEN;
    const bf16* arow = A + (size_t)(m0 + l15) * NROWS + k0 + hi * 8;

#if HAVE_TDM
    unsigned ldsbase = (unsigned)(uintptr_t)&Bt[0][0];
    const unsigned bufbytes = BUFELT * 2;
    // fill buffer `buf` with Yt[0:128][k0+k : k0+k+32] via TDM (wave 0 only)
    #define FILL_B(buf, kk)                                                      \
        do { if (widx == 0) {                                                    \
            unsigned long long ga = (unsigned long long)(uintptr_t)(Yt + k0 + (kk)); \
            u32x4 g0;                                                            \
            g0[0] = 1u;                                /* count=1 */             \
            g0[1] = ldsbase + (unsigned)(buf) * bufbytes;                        \
            g0[2] = (unsigned)ga;                                                \
            g0[3] = ((unsigned)(ga >> 32) & 0x01FFFFFFu) | (2u << 30);           \
            i32x8 g1;                                                            \
            g1[0] = (int)((1u << 16) | (1u << 20) | (3u << 22) | (3u << 25));    \
            g1[1] = (int)(((unsigned)NROWS & 0xFFFFu) << 16);                    \
            g1[2] = (int)(((unsigned)NROWS >> 16) | ((unsigned)NHIDC << 16));    \
            g1[3] = (int)((unsigned)BK << 16);                                   \
            g1[4] = (int)NHIDC;                                                  \
            g1[5] = (int)NROWS;                                                  \
            g1[6] = 0; g1[7] = 0;                                                \
            i32x4 gz = {0, 0, 0, 0};                                             \
            i32x8 gz8 = {0, 0, 0, 0, 0, 0, 0, 0};                                \
            __builtin_amdgcn_tensor_load_to_lds(g0, g1, gz, gz, gz8, 0);         \
        } } while (0)
    #define FILL_WAIT() do { if (widx == 0) __builtin_amdgcn_s_wait_tensorcnt(0); } while (0)
#else
    // cooperative fallback: 256 threads x 32B each
    int frow = tid >> 1, fhf = tid & 1;
    const bf16* fsrc0 = Yt + (size_t)frow * NROWS + k0 + fhf * 16;
    bf16*       fdst  = &Bt[0][frow * BPAD + fhf * 16];
    #define FILL_B(buf, kk)                                                      \
        do {                                                                     \
            const bf16* s_ = fsrc0 + (kk);                                       \
            v8bf d0_ = *(const v8bf*)s_;                                         \
            v8bf d1_ = *(const v8bf*)(s_ + 8);                                   \
            bf16* d_ = fdst + (buf) * BUFELT;                                    \
            *(v8bf*)d_ = d0_;                                                    \
            *(v8bf*)(d_ + 8) = d1_;                                              \
        } while (0)
    #define FILL_WAIT() do { } while (0)
#endif

    v8f acc[8] = {};

    // consume 4 N-tiles (jbase..jbase+3) of buffer `buf` with A fragment `a`
    auto consume4 = [&](int buf, int jbase, const ABFrag& a) {
        const bf16* lb = &Bt[buf][hi * 8];
        ABFrag b[4];
#pragma unroll
        for (int j = 0; j < 4; ++j) {
            const bf16* bp = lb + ((jbase + j) * 16 + l15) * BPAD;
            b[j].h[0] = *(const v8bf*)bp;
            b[j].h[1] = *(const v8bf*)(bp + 16);
        }
#pragma unroll
        for (int j = 0; j < 4; ++j)
            acc[jbase + j] = __builtin_amdgcn_wmma_f32_16x16x32_bf16(
                false, a.v, false, b[j].v, (short)0, acc[jbase + j], false, false);
    };

    // prologue: buffer 0, A fragment 0
    FILL_B(0, 0);
    ABFrag a0, a1;
    a0.h[0] = *(const v8bf*)(arow);
    a0.h[1] = *(const v8bf*)(arow + 16);
    FILL_WAIT();
    __syncthreads();

    for (int kt = 0; kt < KITERS; kt += 2) {
        int k = kt * 32;
        __builtin_prefetch(arow + k + 256, 0, 1);   // A stream, ~8 iters ahead
        // ---- even iteration: consume buf 0 with a0; fill buf 1; load a1 ----
        FILL_B(1, k + 32);
        a1.h[0] = *(const v8bf*)(arow + k + 32);
        a1.h[1] = *(const v8bf*)(arow + k + 48);
        consume4(0, 0, a0);
        consume4(0, 4, a0);
        FILL_WAIT();
        __syncthreads();
        // ---- odd iteration: consume buf 1 with a1; fill buf 0; load a0 ----
        if (kt + 2 < KITERS) {
            FILL_B(0, k + 64);
            a0.h[0] = *(const v8bf*)(arow + k + 64);
            a0.h[1] = *(const v8bf*)(arow + k + 80);
        }
        consume4(1, 0, a1);
        consume4(1, 4, a1);
        FILL_WAIT();
        __syncthreads();
    }
#undef FILL_B
#undef FILL_WAIT

    int mb = m0 + (hi << 3);
#pragma unroll
    for (int j = 0; j < 8; ++j)
#pragma unroll
        for (int v = 0; v < 8; ++v)
            C[(size_t)(mb + v) * NHIDC + (j << 4) + l15] = acc[j][v];
}

// ---------------------------------------------------------------------------
// Per step: sum the 4 K-chunk partials (deterministic order) and do the four
// 8192x128x128 small GEMMs (f32 WMMA 16x16x4; S/P symmetric so B fragments
// are contiguous row loads) fused with the elementwise update.
// ---------------------------------------------------------------------------
__global__ __launch_bounds__(256) void k_update(const float* __restrict__ Yin,
                                                const float* __restrict__ Y0,
                                                const float* __restrict__ Pab,
                                                const float* __restrict__ Pag,
                                                const float* __restrict__ S1,
                                                const float* __restrict__ P1,
                                                const float* __restrict__ S2,
                                                const float* __restrict__ P2,
                                                const float* __restrict__ db,
                                                const float* __restrict__ dg,
                                                const float* __restrict__ qinv,
                                                float* __restrict__ Yout) {
    const size_t CH = (size_t)NROWS * NHIDC;    // elements per K-chunk partial
    int wave = blockIdx.x * 8 + (threadIdx.x >> 5);  // 0..4095
    int lane = threadIdx.x & 31;
    int mstrip = wave >> 3;
    int n0     = (wave & 7) << 4;
    int m0     = mstrip << 4;
    int l15    = lane & 15;
    int hi     = lane >> 4;
    int mrow   = m0 + l15;      // A-fragment row for this lane
    int nrow   = n0 + l15;      // B-fragment row (== column, S/P symmetric)

    v8f aS1 = {}, aP1 = {}, aS2 = {}, aP2 = {};
#if HAVE_WMMA_F32X4
    {
        int koff = hi * 2;      // 16x16x4 f32 A/B layout: lanes 16-31 hold K=2,3
        const float* pY  = Yin + (size_t)mrow * NHIDC + koff;
        const float* pAb = Pab + (size_t)mrow * NHIDC + koff;
        const float* pAg = Pag + (size_t)mrow * NHIDC + koff;
        const float* pS1 = S1  + (size_t)nrow * NHIDC + koff;
        const float* pP1 = P1  + (size_t)nrow * NHIDC + koff;
        const float* pS2 = S2  + (size_t)nrow * NHIDC + koff;
        const float* pP2 = P2  + (size_t)nrow * NHIDC + koff;
#pragma unroll 4
        for (int k = 0; k < NHIDC; k += 4) {
            v2f fY  = *(const v2f*)(pY + k);
            v2f fAb = *(const v2f*)(pAb + k)          + *(const v2f*)(pAb + CH + k)
                    + *(const v2f*)(pAb + 2 * CH + k) + *(const v2f*)(pAb + 3 * CH + k);
            v2f fAg = *(const v2f*)(pAg + k)          + *(const v2f*)(pAg + CH + k)
                    + *(const v2f*)(pAg + 2 * CH + k) + *(const v2f*)(pAg + 3 * CH + k);
            v2f bS1 = *(const v2f*)(pS1 + k);
            v2f bP1 = *(const v2f*)(pP1 + k);
            v2f bS2 = *(const v2f*)(pS2 + k);
            v2f bP2 = *(const v2f*)(pP2 + k);
            aS1 = __builtin_amdgcn_wmma_f32_16x16x4_f32(false, fAb, false, bS1, (short)0, aS1, false, false);
            aP1 = __builtin_amdgcn_wmma_f32_16x16x4_f32(false, fY,  false, bP1, (short)0, aP1, false, false);
            aS2 = __builtin_amdgcn_wmma_f32_16x16x4_f32(false, fAg, false, bS2, (short)0, aS2, false, false);
            aP2 = __builtin_amdgcn_wmma_f32_16x16x4_f32(false, fY,  false, bP2, (short)0, aP2, false, false);
        }
    }
#else
    // Scalar fallback producing the same C-layout slots.
    for (int v = 0; v < 8; ++v) {
        int m = m0 + v + (hi << 3);
        float s1 = 0.f, p1 = 0.f, s2 = 0.f, p2 = 0.f;
        for (int k = 0; k < NHIDC; ++k) {
            size_t mi = (size_t)m * NHIDC + k;
            float ab = Pab[mi] + Pab[CH + mi] + Pab[2 * CH + mi] + Pab[3 * CH + mi];
            float ag = Pag[mi] + Pag[CH + mi] + Pag[2 * CH + mi] + Pag[3 * CH + mi];
            float yv = Yin[mi];
            s1 = fmaf(ab, S1[(size_t)nrow * NHIDC + k], s1);
            p1 = fmaf(yv, P1[(size_t)nrow * NHIDC + k], p1);
            s2 = fmaf(ag, S2[(size_t)nrow * NHIDC + k], s2);
            p2 = fmaf(yv, P2[(size_t)nrow * NHIDC + k], p2);
        }
        aS1[v] = s1; aP1[v] = p1; aS2[v] = s2; aP2[v] = p2;
    }
#endif
    // Elementwise finish directly in the WMMA C layout: slot (v) -> row mb+v, col nrow
    int mb = m0 + (hi << 3);
#pragma unroll
    for (int v = 0; v < 8; ++v) {
        int m = mb + v;
        size_t idx = (size_t)m * NHIDC + nrow;
        float y   = Yin[idx];
        float y0v = Y0[idx];
        float agy = Pag[idx] + Pag[CH + idx] + Pag[2 * CH + idx] + Pag[3 * CH + idx];
        float dbm = db[m];
        float dgm = dg[m];
        float yhat = LAM0C * (aS1[v] - dbm * aP1[v]) + y0v
                   + LAM1C * (dgm * y - agy + aS2[v] - dgm * aP2[v]);
        Yout[idx] = (1.0f - ALPHAC) * y + ALPHAC * (qinv[m] * yhat);
    }
}

// ---------------------------------------------------------------------------
// Epilogue: out = relu(Y) @ W2^T + b2   (8192 x 40, negligible cost)
// ---------------------------------------------------------------------------
__global__ __launch_bounds__(256) void k_out(const float* __restrict__ Y,
                                             const float* __restrict__ W2,
                                             const float* __restrict__ b2,
                                             float* __restrict__ out) {
    int gid = blockIdx.x * 256 + threadIdx.x;
    if (gid >= NROWS * NCLASSC) return;
    int m = gid / NCLASSC, c = gid % NCLASSC;
    const float* yr = Y  + (size_t)m * NHIDC;
    const float* wr = W2 + (size_t)c * NHIDC;
    float acc = b2[c];
    for (int k = 0; k < NHIDC; ++k) acc = fmaf(fmaxf(yr[k], 0.0f), wr[k], acc);
    out[gid] = acc;
}

// ---------------------------------------------------------------------------
extern "C" void kernel_launch(void* const* d_in, const int* in_sizes, int n_in,
                              void* d_out, int out_size, void* d_ws, size_t ws_size,
                              hipStream_t stream) {
    const float* x       = (const float*)d_in[0];
    const float* A_beta  = (const float*)d_in[1];
    const float* A_gamma = (const float*)d_in[2];
    const float* d_beta  = (const float*)d_in[3];
    const float* d_gamma = (const float*)d_in[4];
    const float* H1      = (const float*)d_in[5];
    const float* H2      = (const float*)d_in[6];
    const float* W1      = (const float*)d_in[7];
    const float* b1      = (const float*)d_in[8];
    const float* W2      = (const float*)d_in[9];
    const float* b2      = (const float*)d_in[10];

    char*  ws  = (char*)d_ws;
    size_t off = 0;
    bf16* AbB = (bf16*)(ws + off); off += (size_t)NROWS * NROWS * 2;   // 128 MB
    bf16* AgB = (bf16*)(ws + off); off += (size_t)NROWS * NROWS * 2;   // 128 MB
    bf16* Yt  = (bf16*)(ws + off); off += (size_t)NHIDC * NROWS * 2;   //   2 MB
    float* Y0  = (float*)(ws + off); off += (size_t)NROWS * NHIDC * 4; //   4 MB
    float* Yb0 = (float*)(ws + off); off += (size_t)NROWS * NHIDC * 4;
    float* Yb1 = (float*)(ws + off); off += (size_t)NROWS * NHIDC * 4;
    float* Pab = (float*)(ws + off); off += (size_t)KCHUNKS * NROWS * NHIDC * 4; // 16 MB
    float* Pag = (float*)(ws + off); off += (size_t)KCHUNKS * NROWS * NHIDC * 4; // 16 MB
    float* S1  = (float*)(ws + off); off += (size_t)NHIDC * NHIDC * 4;
    float* P1  = (float*)(ws + off); off += (size_t)NHIDC * NHIDC * 4;
    float* S2  = (float*)(ws + off); off += (size_t)NHIDC * NHIDC * 4;
    float* P2  = (float*)(ws + off); off += (size_t)NHIDC * NHIDC * 4;
    float* qinv = (float*)(ws + off); off += (size_t)NROWS * 4;

    k_convertA<<<65536, 256, 0, stream>>>(A_beta, A_gamma, AbB, AgB);
    k_prep<<<96, 256, 0, stream>>>(H1, H2, d_beta, d_gamma, S1, P1, S2, P2, qinv);
    k_y0<<<4096, 256, 0, stream>>>(x, W1, b1, Y0);

    const float* Yin = Y0;
    for (int s = 0; s < STEPSC; ++s) {
        k_transpose<<<dim3(256, 4), 256, 0, stream>>>(Yin, Yt);
        k_biggemm<<<512, 256, 0, stream>>>(AbB, AgB, Yt, Pab, Pag);
        float* Yout = (s & 1) ? Yb1 : Yb0;
        k_update<<<512, 256, 0, stream>>>(Yin, Y0, Pab, Pag, S1, P1, S2, P2,
                                          d_beta, d_gamma, qinv, Yout);
        Yin = Yout;
    }
    k_out<<<1280, 256, 0, stream>>>(Yin, W2, b2, (float*)d_out);
}
